// GRU_59983513256390
// MI455X (gfx1250) — matrix operand order repaired and checked by reading, not compiled, not run on hardware
//
#include <hip/hip_runtime.h>
#include <hip/hip_bf16.h>

typedef __attribute__((ext_vector_type(16))) __bf16 v16bf;
typedef __attribute__((ext_vector_type(8)))  float  v8f;

#define B_TOT   16384
#define T_STEPS 28
#define F_IN    28
#define H_U     128
#define C_OUT   10
#define TH3     384     // 3*H
#define NT_G    24      // n-tiles covering 3H
#define KT_H    4       // k-tiles covering H (128/32)
#define FRAG    512     // elements per 32x16 B fragment

// d_ws layout (bf16 elements), all weights pre-swizzled to WMMA B-fragment order
#define OFF_BK  0
#define CNT_BK  (NT_G*FRAG)          // 12288  : kernel (28->32 x 384)
#define OFF_BR  (OFF_BK+CNT_BK)      // 12288
#define CNT_BR  (KT_H*NT_G*FRAG)     // 49152  : recurrent (128 x 384)
#define OFF_BD  (OFF_BR+CNT_BR)      // 61440
#define CNT_BD  (KT_H*FRAG)          // 2048   : dense (128 x 10->16)
#define CNT_ALL (OFF_BD+CNT_BD)      // 63488

#define WAVES    4
#define TILE_B   (WAVES*16)          // 64 batch rows per block
#define NBLOCKS  (B_TOT/TILE_B)      // 256

#define WMMA_BF16(A, Bf, Cf) \
  __builtin_amdgcn_wmma_f32_16x16x32_bf16(false, (A), false, (Bf), (short)0, (Cf), false, false)

__device__ __forceinline__ float fast_sigmoid(float x) {
  // sigmoid(x) = 0.5*tanh(x/2) + 0.5 ; v_tanh_f32 is a single TRANS op on gfx1250
  return __builtin_fmaf(0.5f, __builtin_amdgcn_tanhf(0.5f * x), 0.5f);
}

// B fragment element map: flat = lane*16 + e ; k = 8*(lane/16) + (e&7) + 16*(e>>3) ; n = lane%16
__global__ void prep_weights(const float* __restrict__ ker,
                             const float* __restrict__ rk,
                             const float* __restrict__ dw,
                             __bf16* __restrict__ ws) {
  int g = blockIdx.x * blockDim.x + threadIdx.x;
  if (g >= CNT_ALL) return;
  float val = 0.0f;
  if (g < OFF_BR) {                      // input kernel, K padded 28->32
    int idx = g;
    int nt = idx / FRAG, rem = idx % FRAG;
    int lane = rem >> 4, e = rem & 15;
    int k = 8 * (lane >> 4) + (e & 7) + 16 * (e >> 3);
    int n = 16 * nt + (lane & 15);
    if (k < F_IN) val = ker[k * TH3 + n];
  } else if (g < OFF_BD) {               // recurrent kernel
    int idx = g - OFF_BR;
    int frag = idx / FRAG, rem = idx % FRAG;
    int kt = frag / NT_G, nt = frag % NT_G;
    int lane = rem >> 4, e = rem & 15;
    int k = 32 * kt + 8 * (lane >> 4) + (e & 7) + 16 * (e >> 3);
    int n = 16 * nt + (lane & 15);
    val = rk[k * TH3 + n];
  } else {                               // dense, N padded 10->16
    int idx = g - OFF_BD;
    int kt = idx / FRAG, rem = idx % FRAG;
    int lane = rem >> 4, e = rem & 15;
    int k = 32 * kt + 8 * (lane >> 4) + (e & 7) + 16 * (e >> 3);
    int n = lane & 15;
    if (n < C_OUT) val = dw[k * C_OUT + n];
  }
  ws[g] = (__bf16)val;
}

union UA { v16bf v; __bf16 e[16]; };
union UF { v8f  v; float  f[8];  };
union UX { float4 q; float f[4]; };

// 15 B fragments for one gate-column group j: [0..2]=Bk(z,r,h), [3..6]=Br z, [7..10]=Br r, [11..14]=Br h
struct FragSet { v16bf a[15]; };

__device__ __forceinline__ void load_frags(FragSet& f, const __bf16* s_w, int j, int lane) {
  f.a[0] = *(const v16bf*)(s_w + OFF_BK + j * FRAG + lane * 16);
  f.a[1] = *(const v16bf*)(s_w + OFF_BK + (8 + j) * FRAG + lane * 16);
  f.a[2] = *(const v16bf*)(s_w + OFF_BK + (16 + j) * FRAG + lane * 16);
  #pragma unroll
  for (int kk = 0; kk < 4; ++kk) {
    f.a[3 + kk]  = *(const v16bf*)(s_w + OFF_BR + (kk * NT_G + j)      * FRAG + lane * 16);
    f.a[7 + kk]  = *(const v16bf*)(s_w + OFF_BR + (kk * NT_G + 8 + j)  * FRAG + lane * 16);
    f.a[11 + kk] = *(const v16bf*)(s_w + OFF_BR + (kk * NT_G + 16 + j) * FRAG + lane * 16);
  }
}

extern __shared__ __align__(128) char smem_raw[];

__global__ __launch_bounds__(128, 1) void gru_main(
    const float* __restrict__ x, const float* __restrict__ bias,
    const float* __restrict__ dense_b, const __bf16* __restrict__ ws,
    float* __restrict__ out)
{
  // LDS map:  weights (bf16) | bias_i,bias_r (f32) | h state (bf16)
  __bf16* s_w  = (__bf16*)smem_raw;                         // CNT_ALL
  float*  s_bi = (float*)(smem_raw + (size_t)CNT_ALL * 2);  // 384
  float*  s_br = s_bi + TH3;                                // 384
  __bf16* s_h  = (__bf16*)(s_br + TH3);                     // WAVES * 16*128

  const int tid = threadIdx.x;

  // ---- stage weights + biases into LDS (16B vector copies), zero h ----
  {
    const uint4* src = (const uint4*)ws;
    uint4* dst = (uint4*)s_w;
    const int n16 = (CNT_ALL * 2) / 16;
    for (int i = tid; i < n16; i += blockDim.x) dst[i] = src[i];
    for (int i = tid; i < 2 * TH3; i += blockDim.x) s_bi[i] = bias[i];
    unsigned* hz = (unsigned*)s_h;
    const int nh = (WAVES * 16 * 128 * 2) / 4;
    for (int i = tid; i < nh; i += blockDim.x) hz[i] = 0u;
  }
  __syncthreads();

  const int wave  = tid >> 5;
  const int lane  = tid & 31;
  const int lhalf = lane >> 4;     // 0 or 1
  const int lcol  = lane & 15;
  const long rowbase = (long)blockIdx.x * TILE_B + wave * 16;

  __bf16* hbuf = s_h + wave * 2048;   // 16 rows x 128 cols, row-major, in-place update

  const v8f zero8 = {0.f,0.f,0.f,0.f,0.f,0.f,0.f,0.f};
  const float* xlane = x + ((long)(rowbase + lcol)) * T_STEPS * F_IN;

  // ---- hoist t-invariant gate biases into registers (32 scalars/lane) ----
  float bZ[8], bR[8], bHx[8], bHr[8];
  #pragma unroll
  for (int j = 0; j < 8; ++j) {
    bZ[j]  = s_bi[16 * j + lcol]        + s_br[16 * j + lcol];
    bR[j]  = s_bi[16 * (8 + j) + lcol]  + s_br[16 * (8 + j) + lcol];
    bHx[j] = s_bi[16 * (16 + j) + lcol];
    bHr[j] = s_br[16 * (16 + j) + lcol];
  }

  // ---- prime the B-fragment software pipeline (fragments are t-invariant) ----
  FragSet f0, f1;
  load_frags(f0, s_w, 0, lane);

  for (int t = 0; t < T_STEPS; ++t) {
    // ---- A fragment for x_t (16x32, K padded): 3-4 float4 loads per lane ----
    UA xa;
    {
      const float* xr = xlane + t * F_IN;
      const int k0 = lhalf * 8;        // 0 or 8   -> elems 0..7  = K k0..k0+7
      const int k1 = 16 + k0;          // 16 or 24 -> elems 8..15 = K k1..k1+7
      UX a0, a1, b0, b1;
      a0.q = *(const float4*)(xr + k0);
      a1.q = *(const float4*)(xr + k0 + 4);
      b0.q = *(const float4*)(xr + k1);
      if (k1 + 4 < F_IN) b1.q = *(const float4*)(xr + k1 + 4);
      else { b1.f[0]=b1.f[1]=b1.f[2]=b1.f[3]=0.f; }
      #pragma unroll
      for (int e = 0; e < 4; ++e) {
        xa.e[e]      = (__bf16)a0.f[e];
        xa.e[4 + e]  = (__bf16)a1.f[e];
        xa.e[8 + e]  = (__bf16)b0.f[e];
        xa.e[12 + e] = (__bf16)b1.f[e];
      }
      if (t + 1 < T_STEPS) __builtin_prefetch(xr + F_IN, 0, 1);  // global_prefetch_b8
    }
    // ---- A fragments for h (4 k-tiles of 16x32) from LDS ----
    UA ha[4];
    #pragma unroll
    for (int kk = 0; kk < 4; ++kk) {
      const __bf16* hrow = hbuf + lcol * 128;
      #pragma unroll
      for (int e = 0; e < 16; ++e) {
        int k = 32 * kk + lhalf * 8 + (e & 7) + (e >> 3) * 16;
        ha[kk].e[e] = hrow[k];
      }
    }

    // ---- 8 gate-column groups, ping-pong pipelined B fragments ----
    #pragma unroll
    for (int j = 0; j < 8; ++j) {
      FragSet& cur = (j & 1) ? f1 : f0;
      FragSet& nxt = (j & 1) ? f0 : f1;
      // issue next group's 30 ds_load_b128 NOW; they retire behind 15 WMMAs
      load_frags(nxt, s_w, (j + 1) & 7, lane);

      UF aZ, aR, aHx, aHr;
      aZ.v  = WMMA_BF16(xa.v, cur.a[0], zero8);
      aR.v  = WMMA_BF16(xa.v, cur.a[1], zero8);
      aHx.v = WMMA_BF16(xa.v, cur.a[2], zero8);
      aHr.v = zero8;
      #pragma unroll
      for (int kk = 0; kk < 4; ++kk) {
        aZ.v  = WMMA_BF16(ha[kk].v, cur.a[3 + kk],  aZ.v);
        aR.v  = WMMA_BF16(ha[kk].v, cur.a[7 + kk],  aR.v);
        aHr.v = WMMA_BF16(ha[kk].v, cur.a[11 + kk], aHr.v);
      }
      // gate nonlinearity on accumulator layout: row = v + 8*lhalf, col = 16*j + lcol
      #pragma unroll
      for (int v = 0; v < 8; ++v) {
        const int row = v + 8 * lhalf;
        const int col = 16 * j + lcol;
        float z  = fast_sigmoid(aZ.f[v] + bZ[j]);
        float r  = fast_sigmoid(aR.f[v] + bR[j]);
        float hh = __builtin_amdgcn_tanhf((aHx.f[v] + bHx[j]) + r * (aHr.f[v] + bHr[j]));
        float ho = (float)hbuf[row * 128 + col];   // read-before-write, same-wave LDS order
        hbuf[row * 128 + col] = (__bf16)(z * ho + (1.0f - z) * hh);
      }
    }
  }

  // ---- dense (H=128 -> C=10 padded to 16) + softmax ----
  {
    UF acc;
    v16bf bd[4];
    UA ha[4];
    #pragma unroll
    for (int kk = 0; kk < 4; ++kk) {
      bd[kk] = *(const v16bf*)(s_w + OFF_BD + kk * FRAG + lane * 16);
      const __bf16* hrow = hbuf + lcol * 128;
      #pragma unroll
      for (int e = 0; e < 16; ++e) {
        int k = 32 * kk + lhalf * 8 + (e & 7) + (e >> 3) * 16;
        ha[kk].e[e] = hrow[k];
      }
    }
    acc.v = zero8;
    #pragma unroll
    for (int kk = 0; kk < 4; ++kk)
      acc.v = WMMA_BF16(ha[kk].v, bd[kk], acc.v);

    float db = (lcol < C_OUT) ? dense_b[lcol] : 0.0f;
    #pragma unroll
    for (int v = 0; v < 8; ++v) {
      float val = (lcol < C_OUT) ? (acc.f[v] + db) : -1e30f;
      float m = val;
      #pragma unroll
      for (int s = 8; s >= 1; s >>= 1) m = fmaxf(m, __shfl_xor(m, s, 16));
      float ex = (lcol < C_OUT) ? __expf(val - m) : 0.0f;
      float sum = ex;
      #pragma unroll
      for (int s = 8; s >= 1; s >>= 1) sum += __shfl_xor(sum, s, 16);
      if (lcol < C_OUT) {
        long rg = rowbase + v + 8 * lhalf;
        out[rg * C_OUT + lcol] = ex * __builtin_amdgcn_rcpf(sum);
      }
    }
  }
}

extern "C" void kernel_launch(void* const* d_in, const int* in_sizes, int n_in,
                              void* d_out, int out_size, void* d_ws, size_t ws_size,
                              hipStream_t stream) {
  const float* x    = (const float*)d_in[0];
  const float* ker  = (const float*)d_in[1];
  const float* rk   = (const float*)d_in[2];
  const float* bias = (const float*)d_in[3];
  const float* dw   = (const float*)d_in[4];
  const float* db   = (const float*)d_in[5];
  float* out = (float*)d_out;
  __bf16* ws = (__bf16*)d_ws;

  // Pre-swizzle weights to WMMA B-fragment layout in workspace
  prep_weights<<<(CNT_ALL + 255) / 256, 256, 0, stream>>>(ker, rk, dw, ws);

  // Dynamic LDS: weights 124KB + biases 3KB + h state 16KB  (~143KB < 320KB/WGP)
  size_t smem = (size_t)CNT_ALL * 2 + (size_t)2 * TH3 * 4 + (size_t)WAVES * 16 * 128 * 2;
  hipFuncSetAttribute((const void*)gru_main,
                      hipFuncAttributeMaxDynamicSharedMemorySize, (int)smem);
  gru_main<<<NBLOCKS, 128, smem, stream>>>(x, bias, db, ws, out);
}
